// Attention_57836029608163
// MI455X (gfx1250) — matrix operand order repaired and checked by reading, not compiled, not run on hardware
//
#include <hip/hip_runtime.h>
#include <hip/hip_bf16.h>
#include <math.h>

typedef _Float16 f16;
typedef __attribute__((ext_vector_type(16))) _Float16 v16h;
typedef __attribute__((ext_vector_type(8)))  _Float16 v8h;
typedef __attribute__((ext_vector_type(8)))  float    v8f;
typedef int v4i_ __attribute__((vector_size(16)));

#define BB 4
#define NN 1024
#define CCH 768
#define HH 8
#define HD 96

#if defined(__has_builtin)
#if __has_builtin(__builtin_amdgcn_global_load_async_to_lds_b128) && \
    __has_builtin(__builtin_amdgcn_s_wait_asynccnt)
#define HAS_ASYNC_LDS 1
#endif
#endif

static __device__ __forceinline__ v16h make_v16(v8h lo, v8h hi) {
  return __builtin_shufflevector(lo, hi, 0,1,2,3,4,5,6,7,8,9,10,11,12,13,14,15);
}

// ---------------- prep kernels ----------------

__global__ void cvt_f32_f16(const float* __restrict__ src, f16* __restrict__ dst, int n) {
  int i = blockIdx.x * blockDim.x + threadIdx.x;
  if (i < n) dst[i] = (f16)src[i];
}

// src [rows][cols] f32  ->  dst [cols][rows] f16
__global__ void transpose_cvt(const float* __restrict__ src, f16* __restrict__ dst,
                              int rows, int cols) {
  int i = blockIdx.x * blockDim.x + threadIdx.x;
  if (i < rows * cols) {
    int r = i / cols, c = i - r * cols;
    dst[c * rows + r] = (f16)src[i];
  }
}

// ---------------- QKV GEMM: [4096x768]f16 @ [768x2304] -> q/k/v f16 ----------------

__global__ void __launch_bounds__(32) qkv_gemm(const f16* __restrict__ xh,
                                               const f16* __restrict__ wt,
                                               f16* __restrict__ qh,
                                               f16* __restrict__ kh,
                                               f16* __restrict__ vt) {
  const int lane  = threadIdx.x;
  const int m0    = blockIdx.x * 16;
  const int c0    = blockIdx.y * 64;
  const int lo16  = lane & 15;
  const int abase = (lane < 16) ? 0 : 16;
  const int bbase = (lane < 16) ? 0 : 16;
  const size_t arow = (size_t)(m0 + lo16) * CCH;

  v8f acc[4] = {};
  for (int ks = 0; ks < CCH; ks += 32) {
    v16h a = make_v16(*(const v8h*)(xh + arow + ks + abase),
                      *(const v8h*)(xh + arow + ks + abase + 8));
#pragma unroll
    for (int s = 0; s < 4; s++) {
      const f16* bp = wt + (size_t)(c0 + s * 16 + lo16) * CCH + ks + bbase;
      v16h b = make_v16(*(const v8h*)bp, *(const v8h*)(bp + 8));
      acc[s] = __builtin_amdgcn_wmma_f32_16x16x32_f16(false, a, false, b,
                                                      (short)0, acc[s], false, false);
    }
  }

  const int rlo = (lane < 16) ? 0 : 8;
#pragma unroll
  for (int s = 0; s < 4; s++) {
    int c = c0 + s * 16 + lo16;
    int which = c / CCH;
    int h = (c % CCH) / HD;
    int d = c % HD;
#pragma unroll
    for (int r = 0; r < 8; r++) {
      int g = m0 + rlo + r;
      int b = g >> 10, n = g & 1023;
      int bh = b * HH + h;
      f16 val = (f16)acc[s][r];
      if (which == 0)      qh[((size_t)bh * NN + n) * HD + d] = val;
      else if (which == 1) kh[((size_t)bh * NN + n) * HD + d] = val;
      else                 vt[((size_t)bh * HD + d) * NN + n] = val;
    }
  }
}

// ---------------- grouped flash attention ----------------
// grid: (N/64, B*H, 3), 128 threads (4 waves). Each wave owns 16 q rows; the
// block shares 32-key K/V tiles staged in LDS (double-buffered, async loads).

#define KT_STRIDE 40   // 32 halfs + 8 pad (80B rows; keeps 16B alignment, spreads banks)

__global__ void __launch_bounds__(128) attn_kernel(const f16* __restrict__ qh,
                                                   const f16* __restrict__ kh,
                                                   const f16* __restrict__ vt,
                                                   f16* __restrict__ attnout) {
  __shared__ f16 ktile[2][32 * KT_STRIDE];
  __shared__ f16 vtile[2][32 * KT_STRIDE];
  __shared__ f16 ptile[4][16 * 32];

  const int tid  = threadIdx.x;
  const int wave = tid >> 5;
  const int lane = tid & 31;
  const int q0 = blockIdx.x * 64 + wave * 16;
  const int bh = blockIdx.y;
  const int g  = blockIdx.z;
  const int qg  = (g == 0) ? 1 : 2;   // q group
  const int ka  = (g == 2) ? 1 : 0;   // first  kv group
  const int kbg = (g == 0) ? 1 : 2;   // second kv group

  const float SCALE = 0.102062072615966f;  // 96^-0.5 (full head dim, like ref)
  const float SL2E  = SCALE * 1.4426950408889634f;

  const int lo16  = lane & 15;
  const int abase = (lane < 16) ? 0 : 16;
  const int bbase = (lane < 16) ? 0 : 16;

  // stage one 32-key K tile + V tile into LDS buffer `buf`
  const int crow = tid >> 2;        // 0..31
  const int cchk = (tid & 3) * 8;   // half offset of 16B chunk
  auto stage = [&](int kt, int buf) {
    const int grp = (kt < 32) ? ka : kbg;
    const int nk  = (kt & 31) * 32;
    const f16* kg = kh + ((size_t)bh * NN + nk + crow) * HD + grp * 32 + cchk;
    const f16* vg = vt + ((size_t)bh * HD + grp * 32 + crow) * NN + nk + cchk;
    f16* kd = &ktile[buf][crow * KT_STRIDE + cchk];
    f16* vd = &vtile[buf][crow * KT_STRIDE + cchk];
#ifdef HAS_ASYNC_LDS
    __builtin_amdgcn_global_load_async_to_lds_b128(
        (__attribute__((address_space(1))) v4i_*)kg,
        (__attribute__((address_space(3))) v4i_*)kd, 0, 0);
    __builtin_amdgcn_global_load_async_to_lds_b128(
        (__attribute__((address_space(1))) v4i_*)vg,
        (__attribute__((address_space(3))) v4i_*)vd, 0, 0);
#else
    *(v8h*)kd = *(const v8h*)kg;
    *(v8h*)vd = *(const v8h*)vg;
#endif
  };

  // Q fragment (A layout, 16x32), held in registers for the whole pass
  const f16* qp = qh + ((size_t)bh * NN + q0 + lo16) * HD + qg * 32 + abase;
  v16h qa = make_v16(*(const v8h*)qp, *(const v8h*)(qp + 8));

  float mrun[8], lrun[8];
  v8f o0 = {}, o1 = {};
#pragma unroll
  for (int r = 0; r < 8; r++) { mrun[r] = -INFINITY; lrun[r] = 0.f; }

  stage(0, 0);
#ifdef HAS_ASYNC_LDS
  __builtin_amdgcn_s_wait_asynccnt(0);
#endif
  __syncthreads();

  for (int kt = 0; kt < 64; kt++) {
    const int cur = kt & 1;
    if (kt + 1 < 64) stage(kt + 1, cur ^ 1);   // prefetch next tile (async)

    // K fragments (B layout): column = key (tile-local), K-dim = d
    const f16* kp0 = &ktile[cur][lo16 * KT_STRIDE + bbase];
    v16h kf0 = make_v16(*(const v8h*)kp0, *(const v8h*)(kp0 + 8));
    const f16* kp1 = &ktile[cur][(16 + lo16) * KT_STRIDE + bbase];
    v16h kf1 = make_v16(*(const v8h*)kp1, *(const v8h*)(kp1 + 8));

    v8f s0 = {}, s1 = {};
    s0 = __builtin_amdgcn_wmma_f32_16x16x32_f16(false, qa, false, kf0, (short)0, s0, false, false);
    s1 = __builtin_amdgcn_wmma_f32_16x16x32_f16(false, qa, false, kf1, (short)0, s1, false, false);

    // online softmax per row (row r in lanes 0-15 / row r+8 in lanes 16-31)
#pragma unroll
    for (int r = 0; r < 8; r++) {
      float a = s0[r] * SL2E;
      float b = s1[r] * SL2E;
      float t = fmaxf(a, b);
#pragma unroll
      for (int off = 8; off >= 1; off >>= 1) t = fmaxf(t, __shfl_xor(t, off, 16));
      float mnew = fmaxf(mrun[r], t);
      float p0 = exp2f(a - mnew);
      float p1 = exp2f(b - mnew);
      float corr = exp2f(mrun[r] - mnew);
      float ps = p0 + p1;
#pragma unroll
      for (int off = 8; off >= 1; off >>= 1) ps += __shfl_xor(ps, off, 16);
      lrun[r] = lrun[r] * corr + ps;
      mrun[r] = mnew;
      o0[r] *= corr;
      o1[r] *= corr;
      int rowm = (lane < 16) ? r : (r + 8);
      ptile[wave][rowm * 32 + lo16]      = (f16)p0;
      ptile[wave][rowm * 32 + 16 + lo16] = (f16)p1;
    }

    // P: C layout -> A layout via per-wave LDS tile (wave-local dscnt dependency)
    const f16* lp = &ptile[wave][lo16 * 32 + abase];
    v16h pa = make_v16(*(const v8h*)lp, *(const v8h*)(lp + 8));

    // V fragments (B layout): column = d, K-dim = key
    const f16* vp0 = &vtile[cur][lo16 * KT_STRIDE + bbase];
    v16h vf0 = make_v16(*(const v8h*)vp0, *(const v8h*)(vp0 + 8));
    const f16* vp1 = &vtile[cur][(16 + lo16) * KT_STRIDE + bbase];
    v16h vf1 = make_v16(*(const v8h*)vp1, *(const v8h*)(vp1 + 8));

    o0 = __builtin_amdgcn_wmma_f32_16x16x32_f16(false, pa, false, vf0, (short)0, o0, false, false);
    o1 = __builtin_amdgcn_wmma_f32_16x16x32_f16(false, pa, false, vf1, (short)0, o1, false, false);

    // next tile ready + everyone done reading `cur` before it is overwritten
#ifdef HAS_ASYNC_LDS
    __builtin_amdgcn_s_wait_asynccnt(0);
#endif
    __syncthreads();
  }

  // normalize + write into concat buffer: channel = g*256 + h*32 + d
  const int h = bh & 7;
  const int b = bh >> 3;
#pragma unroll
  for (int r = 0; r < 8; r++) {
    int rowm = (lane < 16) ? r : (r + 8);
    float inv = 1.0f / lrun[r];
    size_t base = ((size_t)(b * NN + q0 + rowm)) * CCH + g * 256 + h * 32;
    attnout[base + lo16]      = (f16)(o0[r] * inv);
    attnout[base + 16 + lo16] = (f16)(o1[r] * inv);
  }
}

// ---------------- projection GEMM: [4096x768]f16 @ [768x768] + bias -> f32 ----------------

__global__ void __launch_bounds__(32) proj_gemm(const f16* __restrict__ ah,
                                                const f16* __restrict__ wt,
                                                const float* __restrict__ bias,
                                                float* __restrict__ out) {
  const int lane  = threadIdx.x;
  const int m0    = blockIdx.x * 16;
  const int c0    = blockIdx.y * 64;
  const int lo16  = lane & 15;
  const int abase = (lane < 16) ? 0 : 16;
  const int bbase = (lane < 16) ? 0 : 16;
  const size_t arow = (size_t)(m0 + lo16) * CCH;

  v8f acc[4] = {};
  for (int ks = 0; ks < CCH; ks += 32) {
    v16h a = make_v16(*(const v8h*)(ah + arow + ks + abase),
                      *(const v8h*)(ah + arow + ks + abase + 8));
#pragma unroll
    for (int s = 0; s < 4; s++) {
      const f16* bp = wt + (size_t)(c0 + s * 16 + lo16) * CCH + ks + bbase;
      v16h b = make_v16(*(const v8h*)bp, *(const v8h*)(bp + 8));
      acc[s] = __builtin_amdgcn_wmma_f32_16x16x32_f16(false, a, false, b,
                                                      (short)0, acc[s], false, false);
    }
  }

  const int rlo = (lane < 16) ? 0 : 8;
#pragma unroll
  for (int s = 0; s < 4; s++) {
    int c = c0 + s * 16 + lo16;
    float bv = bias[c];
#pragma unroll
    for (int r = 0; r < 8; r++) {
      out[(size_t)(m0 + rlo + r) * CCH + c] = acc[s][r] + bv;
    }
  }
}

// ---------------- launcher ----------------

extern "C" void kernel_launch(void* const* d_in, const int* in_sizes, int n_in,
                              void* d_out, int out_size, void* d_ws, size_t ws_size,
                              hipStream_t stream) {
  const float* x      = (const float*)d_in[0];   // [4,1024,768]
  const float* W_qkv  = (const float*)d_in[1];   // [768,2304]
  const float* W_proj = (const float*)d_in[2];   // [768,768]
  const float* b_proj = (const float*)d_in[3];   // [768]
  float* out = (float*)d_out;

  const int M = BB * NN;                 // 4096
  const int nX    = M * CCH;             // 3145728
  const int nWqkv = CCH * 3 * CCH;       // 1769472
  const int nWprj = CCH * CCH;           // 589824
  const int nQKV  = BB * HH * NN * HD;   // 3145728 per tensor

  f16* base   = (f16*)d_ws;
  f16* xh     = base;
  f16* wqkvt  = xh + nX;
  f16* wprojt = wqkvt + nWqkv;
  f16* qh     = wprojt + nWprj;
  f16* kh     = qh + nQKV;
  f16* vt     = kh + nQKV;
  f16* attnb  = vt + nQKV;

  cvt_f32_f16<<<(nX + 255) / 256, 256, 0, stream>>>(x, xh, nX);
  transpose_cvt<<<(nWqkv + 255) / 256, 256, 0, stream>>>(W_qkv, wqkvt, CCH, 3 * CCH);
  transpose_cvt<<<(nWprj + 255) / 256, 256, 0, stream>>>(W_proj, wprojt, CCH, CCH);

  qkv_gemm<<<dim3(M / 16, (3 * CCH) / 64), 32, 0, stream>>>(xh, wqkvt, qh, kh, vt);

  attn_kernel<<<dim3(NN / 64, BB * HH, 3), 128, 0, stream>>>(qh, kh, vt, attnb);

  proj_gemm<<<dim3(M / 16, CCH / 64), 32, 0, stream>>>(attnb, wprojt, b_proj, out);

  (void)in_sizes; (void)n_in; (void)out_size; (void)ws_size;
}
